// FastAttention_46222438039853
// MI455X (gfx1250) — matrix operand order repaired and checked by reading, not compiled, not run on hardware
//
#include <hip/hip_runtime.h>

#define DD 256
#define SS 4096
#define BB 4
#define BSROWS (BB * SS)   // 16384

typedef __attribute__((ext_vector_type(16))) __bf16 v16bf;
typedef __attribute__((ext_vector_type(8)))  __bf16 v8bf;
typedef __attribute__((ext_vector_type(8)))  float  v8f;
typedef __attribute__((ext_vector_type(4)))  int    v4i;

union BF16x16 { v16bf v; v8bf h[2]; };

__device__ __forceinline__ v8f wmma_bf16(v16bf a, v16bf b, v8f c) {
  // D = A(16x32 bf16) * B(32x16 bf16) + C(16x16 f32)
  return __builtin_amdgcn_wmma_f32_16x16x32_bf16(false, a, false, b,
                                                 (short)0, c, false, false);
}

// ---- CDNA5 async global->LDS path (ASYNCcnt), guarded so compile never breaks
#if __has_builtin(__builtin_amdgcn_global_load_async_to_lds_b128) && \
    __has_builtin(__builtin_amdgcn_s_wait_asynccnt)
#define ASYNC_LDS 1
#else
#define ASYNC_LDS 0
#endif

typedef __attribute__((address_space(1))) v4i* as1_v4i;
typedef __attribute__((address_space(3))) v4i* as3_v4i;

__device__ __forceinline__ void cp16_g2l(__bf16* l, const __bf16* g) {
#if ASYNC_LDS
  __builtin_amdgcn_global_load_async_to_lds_b128(
      (as1_v4i)(void*)g, (as3_v4i)(void*)l, 0, 0);
#else
  *(v8bf*)l = *(const v8bf*)g;
#endif
}
__device__ __forceinline__ void g2l_wait() {
#if ASYNC_LDS
  __builtin_amdgcn_s_wait_asynccnt(0);
#endif
}

// ---------------------------------------------------------------------------
// Kernel 1: Y = act(X @ W + b), X:[16384,256] f32, W:[256,256] f32.
// TRANS==0: bf16 row-major [row][256] (Q, K).
// TRANS==1: bf16 transposed per batch [b][d][s] (V^T for the PV GEMM).
// ---------------------------------------------------------------------------
template<int RELU, int TRANS>
__global__ __launch_bounds__(256)
void proj_kernel(const float* __restrict__ X, const float* __restrict__ W,
                 const float* __restrict__ bias, __bf16* __restrict__ out)
{
  __shared__ __bf16 sX[128 * 40];   // x chunk  [128 rows][K=32 (+8 pad)]
  __shared__ __bf16 sW[256 * 40];   // W chunk  [N=256][K=32 (+8 pad)] col-major

  const int tid  = threadIdx.x;
  const int wave = tid >> 5, lane = tid & 31;
  const int half = lane >> 4, l16 = lane & 15;
  const int row0 = blockIdx.x * 128;

  v8f acc[16];
  #pragma unroll
  for (int i = 0; i < 16; ++i)
    #pragma unroll
    for (int j = 0; j < 8; ++j) acc[i][j] = 0.0f;

  for (int kc = 0; kc < 8; ++kc) {
    #pragma unroll
    for (int i = 0; i < 4; ++i) {
      const int idx = tid + 256 * i;
      const int r = idx >> 3, c4 = idx & 7;
      const float4 v = *(const float4*)(X + (size_t)(row0 + r) * DD + kc * 32 + c4 * 4);
      __bf16* p = &sX[r * 40 + c4 * 4];
      p[0] = (__bf16)v.x; p[1] = (__bf16)v.y; p[2] = (__bf16)v.z; p[3] = (__bf16)v.w;
    }
    #pragma unroll 4
    for (int k = 0; k < 32; ++k)
      sW[tid * 40 + k] = (__bf16)W[(size_t)(kc * 32 + k) * DD + tid];
    __syncthreads();

    BF16x16 a;
    {
      const __bf16* pa = &sX[(wave * 16 + l16) * 40 + half * 8];
      a.h[0] = *(const v8bf*)(pa);
      a.h[1] = *(const v8bf*)(pa + 16);
    }
    #pragma unroll
    for (int nt = 0; nt < 16; ++nt) {
      BF16x16 bm;
      const __bf16* pb = &sW[(nt * 16 + l16) * 40 + half * 8];
      bm.h[0] = *(const v8bf*)(pb);
      bm.h[1] = *(const v8bf*)(pb + 16);
      acc[nt] = wmma_bf16(a.v, bm.v, acc[nt]);
    }
    __syncthreads();
  }

  const int gm0 = row0 + wave * 16 + half * 8;
  #pragma unroll
  for (int nt = 0; nt < 16; ++nt) {
    const int n = nt * 16 + l16;
    const float bn = bias[n];
    if (TRANS == 0) {
      #pragma unroll
      for (int r = 0; r < 8; ++r) {
        float v = acc[nt][r] + bn;
        if (RELU) v = fmaxf(v, 0.0f);
        out[(size_t)(gm0 + r) * DD + n] = (__bf16)v;
      }
    } else {
      const int b = gm0 / SS, s = gm0 % SS;
      v8bf pack;
      #pragma unroll
      for (int r = 0; r < 8; ++r) pack[r] = (__bf16)(acc[nt][r] + bn);
      *(v8bf*)(out + (size_t)b * DD * SS + (size_t)n * SS + s) = pack;
    }
  }
}

// ---------------------------------------------------------------------------
// Kernel 2: flash attention, 4 waves = 64 q rows per block, grid (S/64, B).
// Double-buffered K/V tiles staged with async global->LDS, overlapped with
// the WMMA compute of the current 64-key block. Fragment loads are 2-deep
// software-pipelined in registers so DS latency hides behind WMMA groups.
// ---------------------------------------------------------------------------
#define SK_STRIDE 264                 // 256 + 8 pad (bf16 units)
#define SV_STRIDE 72                  // 64 + 8 pad
#define SP_STRIDE 72
#define SMEM_K    (64 * SK_STRIDE)
#define SMEM_V    (256 * SV_STRIDE)
#define SMEM_BUF  (SMEM_K + SMEM_V)
#define SMEM_P    (4 * 16 * SP_STRIDE)
#define SMEM_BYTES ((2 * SMEM_BUF + SMEM_P) * sizeof(__bf16))

__device__ __forceinline__ void stage_tiles(const __bf16* __restrict__ K,
                                            const __bf16* __restrict__ Vt,
                                            __bf16* sK, __bf16* sV,
                                            int b, int key0, int tid)
{
  {  // K tile [64 keys][256 d] bf16: 128 threads x 16 x 16B
    const int kr = tid >> 1, cc = (tid & 1) * 128;
    const __bf16* gk = K + (size_t)(b * SS + key0 + kr) * DD + cc;
    __bf16* lk = sK + kr * SK_STRIDE + cc;
    #pragma unroll
    for (int j = 0; j < 16; ++j) cp16_g2l(lk + j * 8, gk + j * 8);
  }
  {  // V^T tile [256 d][64 keys] bf16: 128 threads x 2 rows x 8 x 16B
    #pragma unroll
    for (int rr = 0; rr < 2; ++rr) {
      const int d = tid + rr * 128;
      const __bf16* gv = Vt + (size_t)b * DD * SS + (size_t)d * SS + key0;
      __bf16* lv = sV + d * SV_STRIDE;
      #pragma unroll
      for (int j = 0; j < 8; ++j) cp16_g2l(lv + j * 8, gv + j * 8);
    }
  }
}

__global__ __launch_bounds__(128)
void attn_kernel(const __bf16* __restrict__ Q, const __bf16* __restrict__ K,
                 const __bf16* __restrict__ Vt, float* __restrict__ Out)
{
  extern __shared__ __bf16 smem[];
  __bf16* sP = smem + 2 * SMEM_BUF;          // per-wave [16 q][64 keys (+pad)]

  const int tid  = threadIdx.x;
  const int wave = tid >> 5, lane = tid & 31;
  const int half = lane >> 4, l16 = lane & 15;
  const int b  = blockIdx.y;
  const int q0 = blockIdx.x * 64;

  // Q fragments for this wave's 16 rows, all of D=256 (64 VGPRs)
  BF16x16 qf[8];
  {
    const __bf16* qp = Q + (size_t)(b * SS + q0 + wave * 16 + l16) * DD;
    #pragma unroll
    for (int kc = 0; kc < 8; ++kc) {
      qf[kc].h[0] = *(const v8bf*)(qp + kc * 32 + half * 8);
      qf[kc].h[1] = *(const v8bf*)(qp + kc * 32 + half * 8 + 16);
    }
  }

  v8f O[16];
  #pragma unroll
  for (int i = 0; i < 16; ++i)
    #pragma unroll
    for (int j = 0; j < 8; ++j) O[i][j] = 0.0f;
  float rowM[8], rowL[8];
  #pragma unroll
  for (int r = 0; r < 8; ++r) { rowM[r] = -__builtin_inff(); rowL[r] = 0.0f; }

  // prologue: stage first key block into buffer 0
  stage_tiles(K, Vt, smem, smem + SMEM_K, b, 0, tid);

  for (int kb = 0; kb < SS / 64; ++kb) {
    const int cur = kb & 1;
    __bf16* sK = smem + cur * SMEM_BUF;
    __bf16* sV = sK + SMEM_K;

    g2l_wait();           // my async copies into buffer `cur` are done
    __syncthreads();      // everyone's are; everyone finished reading buf `1-cur`

    if (kb + 1 < SS / 64) // overlap next stage with this block's WMMAs
      stage_tiles(K, Vt, smem + (1 - cur) * SMEM_BUF,
                  smem + (1 - cur) * SMEM_BUF + SMEM_K, b, (kb + 1) * 64, tid);

    // ---- scores S = Q @ K^T : 2-deep pipelined fragment groups ----
    v8f sc[4];
    #pragma unroll
    for (int nt = 0; nt < 4; ++nt)
      #pragma unroll
      for (int j = 0; j < 8; ++j) sc[nt][j] = 0.0f;

    BF16x16 kf[2][4];
    #pragma unroll
    for (int nt = 0; nt < 4; ++nt) {
      const __bf16* pb = &sK[(nt * 16 + l16) * SK_STRIDE + half * 8];
      kf[0][nt].h[0] = *(const v8bf*)(pb);
      kf[0][nt].h[1] = *(const v8bf*)(pb + 16);
    }
    #pragma unroll
    for (int kc = 0; kc < 8; ++kc) {
      const int cb = kc & 1;
      if (kc < 7) {   // issue next group's 8 loads before this group's WMMAs
        #pragma unroll
        for (int nt = 0; nt < 4; ++nt) {
          const __bf16* pb =
              &sK[(nt * 16 + l16) * SK_STRIDE + (kc + 1) * 32 + half * 8];
          kf[1 - cb][nt].h[0] = *(const v8bf*)(pb);
          kf[1 - cb][nt].h[1] = *(const v8bf*)(pb + 16);
        }
      }
      #pragma unroll
      for (int nt = 0; nt < 4; ++nt)
        sc[nt] = wmma_bf16(qf[kc].v, kf[cb][nt].v, sc[nt]);
    }

    // ---- online softmax (row stats replicated within each 16-lane half) ----
    float newM[8], scale[8];
    #pragma unroll
    for (int r = 0; r < 8; ++r) {
      float m = fmaxf(fmaxf(sc[0][r], sc[1][r]), fmaxf(sc[2][r], sc[3][r]));
      #pragma unroll
      for (int off = 8; off > 0; off >>= 1) m = fmaxf(m, __shfl_xor(m, off, 32));
      newM[r]  = fmaxf(rowM[r], m);
      scale[r] = __expf(rowM[r] - newM[r]);
      rowM[r]  = newM[r];
    }
    float psum[8];
    #pragma unroll
    for (int r = 0; r < 8; ++r) psum[r] = 0.0f;
    #pragma unroll
    for (int nt = 0; nt < 4; ++nt)
      #pragma unroll
      for (int r = 0; r < 8; ++r) {
        const float p = __expf(sc[nt][r] - newM[r]);
        sc[nt][r] = p;
        psum[r] += p;
      }
    #pragma unroll
    for (int r = 0; r < 8; ++r) {
      float s = psum[r];
      #pragma unroll
      for (int off = 8; off > 0; off >>= 1) s += __shfl_xor(s, off, 32);
      rowL[r] = rowL[r] * scale[r] + s;
    }
    #pragma unroll
    for (int ft = 0; ft < 16; ++ft)
      #pragma unroll
      for (int r = 0; r < 8; ++r) O[ft][r] *= scale[r];

    // ---- P: C-layout -> A-layout via per-wave LDS bounce ----
    __bf16* pw = &sP[wave * 16 * SP_STRIDE];
    #pragma unroll
    for (int nt = 0; nt < 4; ++nt)
      #pragma unroll
      for (int r = 0; r < 8; ++r)
        pw[(r + half * 8) * SP_STRIDE + nt * 16 + l16] = (__bf16)sc[nt][r];
    BF16x16 pf[2];
    #pragma unroll
    for (int kc = 0; kc < 2; ++kc) {
      const __bf16* pp = &pw[l16 * SP_STRIDE + kc * 32 + half * 8];
      pf[kc].h[0] = *(const v8bf*)(pp);
      pf[kc].h[1] = *(const v8bf*)(pp + 16);
    }

    // ---- O += P @ V : 2-deep pipelined B fragments, 16 indep chains ----
    #pragma unroll
    for (int kc = 0; kc < 2; ++kc) {
      BF16x16 vf[2];
      {
        const __bf16* pv = &sV[l16 * SV_STRIDE + kc * 32 + half * 8];
        vf[0].h[0] = *(const v8bf*)(pv);
        vf[0].h[1] = *(const v8bf*)(pv + 16);
      }
      #pragma unroll
      for (int ft = 0; ft < 16; ++ft) {
        const int cb = ft & 1;
        if (ft < 15) {
          const __bf16* pv =
              &sV[((ft + 1) * 16 + l16) * SV_STRIDE + kc * 32 + half * 8];
          vf[1 - cb].h[0] = *(const v8bf*)(pv);
          vf[1 - cb].h[1] = *(const v8bf*)(pv + 16);
        }
        O[ft] = wmma_bf16(pf[kc].v, vf[cb].v, O[ft]);
      }
    }
  }

  // ---- epilogue: O / l, f32 store ----
  #pragma unroll
  for (int r = 0; r < 8; ++r) rowL[r] = 1.0f / rowL[r];
  const int qrow = q0 + wave * 16 + half * 8;
  float* obase = Out + ((size_t)b * SS + qrow) * DD + l16;
  #pragma unroll
  for (int ft = 0; ft < 16; ++ft)
    #pragma unroll
    for (int r = 0; r < 8; ++r)
      obase[(size_t)r * DD + ft * 16] = O[ft][r] * rowL[r];
}

// ---------------------------------------------------------------------------
extern "C" void kernel_launch(void* const* d_in, const int* in_sizes, int n_in,
                              void* d_out, int out_size, void* d_ws, size_t ws_size,
                              hipStream_t stream) {
  (void)in_sizes; (void)n_in; (void)out_size; (void)ws_size;
  const float* x  = (const float*)d_in[0];
  const float* Wq = (const float*)d_in[1];
  const float* bq = (const float*)d_in[2];
  const float* Wk = (const float*)d_in[3];
  const float* bk = (const float*)d_in[4];
  const float* Wv = (const float*)d_in[5];
  const float* bv = (const float*)d_in[6];

  __bf16* Qb  = (__bf16*)d_ws;                       // [16384][256] bf16
  __bf16* Kb  = Qb + (size_t)BSROWS * DD;            // [16384][256] bf16
  __bf16* Vtb = Kb + (size_t)BSROWS * DD;            // [4][256][4096] bf16 (V^T)

  proj_kernel<1, 0><<<dim3(BSROWS / 128), 256, 0, stream>>>(x, Wq, bq, Qb);
  proj_kernel<1, 0><<<dim3(BSROWS / 128), 256, 0, stream>>>(x, Wk, bk, Kb);
  proj_kernel<0, 1><<<dim3(BSROWS / 128), 256, 0, stream>>>(x, Wv, bv, Vtb);

  (void)hipFuncSetAttribute((const void*)attn_kernel,
                            hipFuncAttributeMaxDynamicSharedMemorySize,
                            (int)SMEM_BYTES);
  attn_kernel<<<dim3(SS / 64, BB), 128, SMEM_BYTES, stream>>>(
      Qb, Kb, Vtb, (float*)d_out);
}
